// Netsim_34608846471385
// MI455X (gfx1250) — matrix operand order repaired and checked by reading, not compiled, not run on hardware
//
#include <hip/hip_runtime.h>
#include <hip/hip_bf16.h>
#include <stdint.h>
#include <stddef.h>

// ---------------- constants matching the reference ----------------
#define NNODES 20000
#define NEDGES 320000
#define FHID   128          // Fout of every GAT layer
#define NHEAD  4

typedef __attribute__((ext_vector_type(16))) __bf16 v16bf;
typedef __attribute__((ext_vector_type(8)))  __bf16 v8bf;
typedef __attribute__((ext_vector_type(8)))  float  v8f;

// ---------------- small utility kernels ----------------
__global__ void cvt_bf16(const float* __restrict__ in, __bf16* __restrict__ out, size_t n) {
  size_t i = (size_t)blockIdx.x * blockDim.x + threadIdx.x;
  if (i < n) out[i] = (__bf16)in[i];
}

__global__ void fill_f32(float* __restrict__ p, float v, size_t n) {
  size_t i = (size_t)blockIdx.x * blockDim.x + threadIdx.x;
  if (i < n) p[i] = v;
}

// ---------------- pack weights into WMMA B-fragment order -------------------
// out[mat][kblk][ntile][lane][i] = (bf16) W[mat][kblk*32 + (lane>>4)*16 + i][ntile*16 + (lane&15)]
// so in the GEMM each lane reads its 16 B elements as ONE contiguous 32B load.
__global__ void pack_bfrag(const float* __restrict__ W, __bf16* __restrict__ out,
                           int K, int Nout, int nmat) {
  size_t total = (size_t)nmat * K * Nout;
  size_t idx = (size_t)blockIdx.x * blockDim.x + threadIdx.x;
  if (idx >= total) return;
  int i    = (int)(idx & 15);
  int lane = (int)((idx >> 4) & 31);
  size_t rest = idx >> 9;                 // [mat][kblk][ntile]
  int ntiles = Nout >> 4;
  int ntile = (int)(rest % ntiles); rest /= ntiles;
  int kblks = K >> 5;
  int kblk = (int)(rest % kblks);
  int mat  = (int)(rest / kblks);
  int k    = kblk * 32 + (lane >> 4) * 16 + i;
  int ncol = ntile * 16 + (lane & 15);
  out[idx] = (__bf16)W[((size_t)mat * K + k) * Nout + ncol];
}

// ---------------- WMMA GEMM: C[M,No] = A[M,K] @ Bpacked, bf16 in, f32 out ----
// One wave handles a 32x16 C slab (two 16x16 WMMA tiles sharing one B frag).
// Requires M%32==0, K%32==0, No%16==0 (true here: M=20000, K in {128,256}, No=128),
// so EXEC stays all-ones as WMMA requires.
// blockIdx.z = batch (head); Bp/C advance by strideB/strideC elements.
__global__ __launch_bounds__(32)
void wmma_gemm_bf16(const __bf16* __restrict__ A, const __bf16* __restrict__ Bp,
                    float* __restrict__ C, int K, int Nout,
                    size_t strideB, size_t strideC) {
  const int lane = threadIdx.x & 31;
  const int half = lane >> 4;   // 0: lanes 0-15, 1: lanes 16-31
  const int l    = lane & 15;
  const int n0   = blockIdx.x * 16;
  const int m0   = blockIdx.y * 32;

  const __bf16* Bb = Bp + (size_t)blockIdx.z * strideB;
  float*        Cb = C  + (size_t)blockIdx.z * strideC;

  const __bf16* arow0 = A + (size_t)(m0 + l) * K;        // tile 0 rows
  const __bf16* arow1 = arow0 + (size_t)16 * K;          // tile 1 rows

  const int ntiles = Nout >> 4;
  const __bf16* bfrag = Bb + (size_t)blockIdx.x * 512 + lane * 16;
  const size_t  bstep = (size_t)ntiles * 512;            // next k-block, same ntile

  v8f acc0 = {}, acc1 = {};
  for (int k0 = 0; k0 < K; k0 += 32) {
    // A fragment (contiguous, matching packed B): elems 0..15 -> K=k0+16h..k0+16h+15
    const int ka = k0 + half * 16;
    union { v16bf v; v8bf h[2]; } a0, a1;
    a0.h[0] = *(const v8bf*)(arow0 + ka);
    a0.h[1] = *(const v8bf*)(arow0 + ka + 8);
    a1.h[0] = *(const v8bf*)(arow1 + ka);
    a1.h[1] = *(const v8bf*)(arow1 + ka + 8);

    // Packed B fragment: one contiguous 32B load per lane.
    v16bf b = *(const v16bf*)bfrag;
    bfrag += bstep;

    __builtin_prefetch(arow0 + ka + 32, 0, 1);           // global_prefetch path
    __builtin_prefetch(arow1 + ka + 32, 0, 1);

    acc0 = __builtin_amdgcn_wmma_f32_16x16x32_bf16(
        false, a0.v, false, b, (short)0, acc0, false, false);
    acc1 = __builtin_amdgcn_wmma_f32_16x16x32_bf16(
        false, a1.v, false, b, (short)0, acc1, false, false);
  }
  // C layout: VGPR r -> row m0 + r + 8*half, col n0 + l
#pragma unroll
  for (int r = 0; r < 8; ++r)
    Cb[(size_t)(m0 + r + half * 8) * Nout + n0 + l] = acc0[r];
#pragma unroll
  for (int r = 0; r < 8; ++r)
    Cb[(size_t)(m0 + 16 + r + half * 8) * Nout + n0 + l] = acc1[r];
}

// ---------------- attention scores: s[n][h] = <h[h][n][:], a[h][:]> ----------
__global__ void att_scores(const float* __restrict__ h,
                           const float* __restrict__ a_s,
                           const float* __restrict__ a_d,
                           float* __restrict__ ssrc, float* __restrict__ sdst,
                           int n) {
  int idx = blockIdx.x * blockDim.x + threadIdx.x;   // n*NHEAD
  if (idx >= n * NHEAD) return;
  int node = idx >> 2, hd = idx & 3;
  const float* hp  = h + ((size_t)hd * n + node) * FHID;
  const float* as_ = a_s + hd * FHID;
  const float* ad_ = a_d + hd * FHID;
  float ss = 0.f, sd = 0.f;
#pragma unroll 4
  for (int o = 0; o < FHID; ++o) { float v = hp[o]; ss += v * as_[o]; sd += v * ad_[o]; }
  ssrc[idx] = ss; sdst[idx] = sd;
}

// ---------------- float atomicMax via int/uint monotonicity ----------------
__device__ inline void atomicMaxF(float* addr, float value) {
  if (__float_as_int(value) >= 0)
    atomicMax((int*)addr, __float_as_int(value));
  else
    atomicMin((unsigned int*)addr, __float_as_uint(value));
}

// ---------------- edge pass A: leaky_relu score + segment max over src ------
__global__ void edge_scores(const int* __restrict__ src, const int* __restrict__ dst,
                            const float* __restrict__ ssrc, const float* __restrict__ sdst,
                            float* __restrict__ e, float* __restrict__ m, int E) {
  int idx = blockIdx.x * blockDim.x + threadIdx.x;   // E*NHEAD
  if (idx >= E * NHEAD) return;
  int ed = idx >> 2, hd = idx & 3;
  int s = src[ed], d = dst[ed];
  float v = ssrc[s * NHEAD + hd] + sdst[d * NHEAD + hd];
  v = v > 0.f ? v : 0.2f * v;                        // leaky_relu(0.2)
  e[idx] = v;
  atomicMaxF(&m[s * NHEAD + hd], v);
}

// ---------------- edge pass B: exp(e - m[src]) + segment sum ----------------
__global__ void edge_norm(const int* __restrict__ src, float* __restrict__ e,
                          const float* __restrict__ m, float* __restrict__ den, int E) {
  int idx = blockIdx.x * blockDim.x + threadIdx.x;
  if (idx >= E * NHEAD) return;
  int ed = idx >> 2, hd = idx & 3;
  int s = src[ed];
  float ex = __expf(e[idx] - m[s * NHEAD + hd]);
  e[idx] = ex;
  atomicAdd(&den[s * NHEAD + hd], ex);
}

// ---------------- edge pass C: alpha-weighted message scatter ----------------
// one block per edge, one thread per output feature; reduce heads in registers
// -> exactly one atomicAdd per (edge, feature) to the L2-resident agg buffer.
__global__ __launch_bounds__(FHID)
void edge_msg(const int* __restrict__ src, const int* __restrict__ dst,
              const float* __restrict__ ex, const float* __restrict__ den,
              const float* __restrict__ h, float* __restrict__ agg, int n) {
  int ed = blockIdx.x;
  int o  = threadIdx.x;
  int s = src[ed], d = dst[ed];
  float acc = 0.f;
#pragma unroll
  for (int hd = 0; hd < NHEAD; ++hd) {
    float alpha = ex[ed * NHEAD + hd] / (den[s * NHEAD + hd] + 1e-16f);
    acc += alpha * h[((size_t)hd * n + d) * FHID + o];
  }
  atomicAdd(&agg[(size_t)s * FHID + o], acc);
}

// ---------------- layer finish: elu(agg/H + residual) -----------------------
__global__ void layer_finish(const float* __restrict__ agg, const float* __restrict__ res,
                             float* __restrict__ out, size_t total) {
  size_t i = (size_t)blockIdx.x * blockDim.x + threadIdx.x;
  if (i >= total) return;
  float v = agg[i] * (1.0f / NHEAD) + res[i];
  out[i] = v > 0.f ? v : (__expf(v) - 1.0f);         // ELU(alpha=1)
}

// ---------------- SE block: per-channel mean --------------------------------
__global__ void chan_reduce(const float* __restrict__ x1, const float* __restrict__ x2,
                            const float* __restrict__ d1, const float* __restrict__ d2,
                            float* __restrict__ sums, size_t total) {
  const float* ptrs[4] = {x1, x2, d1, d2};
  const float* p = ptrs[blockIdx.y];
  __shared__ float red[256];
  float s = 0.f;
  for (size_t i = (size_t)blockIdx.x * blockDim.x + threadIdx.x; i < total;
       i += (size_t)gridDim.x * blockDim.x)
    s += p[i];
  red[threadIdx.x] = s;
  __syncthreads();
  for (int off = 128; off > 0; off >>= 1) {
    if ((int)threadIdx.x < off) red[threadIdx.x] += red[threadIdx.x + off];
    __syncthreads();
  }
  if (threadIdx.x == 0) atomicAdd(&sums[blockIdx.y], red[0]);
}

// ---------------- SE FC (4 -> 20 -> 4, sigmoid) ------------------------------
__global__ void se_fc(const float* __restrict__ sums, float inv_total,
                      const float* __restrict__ fc1w, const float* __restrict__ fc1b,
                      const float* __restrict__ fc2w, const float* __restrict__ fc2b,
                      float* __restrict__ att) {
  if (threadIdx.x == 0 && blockIdx.x == 0) {
    float avg[4];
    for (int c = 0; c < 4; ++c) avg[c] = sums[c] * inv_total;
    float hb[20];
    for (int j = 0; j < 20; ++j) {
      float v = fc1b[j];
      for (int c = 0; c < 4; ++c) v += avg[c] * fc1w[c * 20 + j];
      hb[j] = v;
    }
    for (int c = 0; c < 4; ++c) {
      float v = fc2b[c];
      for (int j = 0; j < 20; ++j) v += hb[j] * fc2w[j * 4 + c];
      att[c] = 1.0f / (1.0f + __expf(-v));
    }
  }
}

// ---------------- final combine: sum_c conv_w[c]*relu(att[c]*xc) + conv_b ----
__global__ void se_combine(const float* __restrict__ x1, const float* __restrict__ x2,
                           const float* __restrict__ d1, const float* __restrict__ d2,
                           const float* __restrict__ att, const float* __restrict__ convw,
                           const float* __restrict__ convb,
                           float* __restrict__ out, size_t total) {
  size_t i = (size_t)blockIdx.x * blockDim.x + threadIdx.x;
  if (i >= total) return;
  float r = convb[0];
  r += convw[0] * fmaxf(0.f, att[0] * x1[i]);
  r += convw[1] * fmaxf(0.f, att[1] * x2[i]);
  r += convw[2] * fmaxf(0.f, att[2] * d1[i]);
  r += convw[3] * fmaxf(0.f, att[3] * d2[i]);
  out[i] = r;
}

// ---------------- host-side orchestration ----------------
static inline char* bump(char*& p, size_t bytes) {
  char* r = p;
  p += (bytes + 255) & ~(size_t)255;
  return r;
}

struct LayerWS {
  __bf16 *xb, *wb, *rb;
  float *h, *res, *ssrc, *sdst, *m, *den, *ebuf, *agg;
};

static void gat_layer(const float* X, int Fin,
                      const float* W, const float* as_, const float* ad_, const float* R,
                      const int* src, const int* dst,
                      float* out, const LayerWS& w, hipStream_t stream) {
  const int Nn = NNODES;
  size_t nx = (size_t)Nn * Fin;
  cvt_bf16<<<(unsigned)((nx + 255) / 256), 256, 0, stream>>>(X, w.xb, nx);
  size_t nw = (size_t)NHEAD * Fin * FHID;
  pack_bfrag<<<(unsigned)((nw + 255) / 256), 256, 0, stream>>>(W, w.wb, Fin, FHID, NHEAD);
  size_t nr = (size_t)Fin * FHID;
  pack_bfrag<<<(unsigned)((nr + 255) / 256), 256, 0, stream>>>(R, w.rb, Fin, FHID, 1);

  // h[hd] = X @ W[hd]   (batched over heads; 32x16 C slab per wave)
  dim3 gh(FHID / 16, Nn / 32, NHEAD);
  wmma_gemm_bf16<<<gh, 32, 0, stream>>>(w.xb, w.wb, w.h, Fin, FHID,
                                        (size_t)Fin * FHID, (size_t)Nn * FHID);
  // res = X @ R
  dim3 gr(FHID / 16, Nn / 32, 1);
  wmma_gemm_bf16<<<gr, 32, 0, stream>>>(w.xb, w.rb, w.res, Fin, FHID, 0, 0);

  int nh = Nn * NHEAD;
  att_scores<<<(nh + 255) / 256, 256, 0, stream>>>(w.h, as_, ad_, w.ssrc, w.sdst, Nn);

  fill_f32<<<(nh + 255) / 256, 256, 0, stream>>>(w.m, -1e30f, (size_t)nh);
  fill_f32<<<(nh + 255) / 256, 256, 0, stream>>>(w.den, 0.f, (size_t)nh);
  size_t na = (size_t)Nn * FHID;
  fill_f32<<<(unsigned)((na + 255) / 256), 256, 0, stream>>>(w.agg, 0.f, na);

  int eh = NEDGES * NHEAD;
  edge_scores<<<(eh + 255) / 256, 256, 0, stream>>>(src, dst, w.ssrc, w.sdst, w.ebuf, w.m, NEDGES);
  edge_norm<<<(eh + 255) / 256, 256, 0, stream>>>(src, w.ebuf, w.m, w.den, NEDGES);
  edge_msg<<<NEDGES, FHID, 0, stream>>>(src, dst, w.ebuf, w.den, w.h, w.agg, Nn);

  layer_finish<<<(unsigned)((na + 255) / 256), 256, 0, stream>>>(w.agg, w.res, out, na);
}

extern "C" void kernel_launch(void* const* d_in, const int* in_sizes, int n_in,
                              void* d_out, int out_size, void* d_ws, size_t ws_size,
                              hipStream_t stream) {
  (void)in_sizes; (void)n_in; (void)out_size; (void)ws_size;

  const float* x    = (const float*)d_in[0];
  const float* dat  = (const float*)d_in[1];
  const float* W1   = (const float*)d_in[2];
  const float* a1s  = (const float*)d_in[3];
  const float* a1d  = (const float*)d_in[4];
  const float* R1   = (const float*)d_in[5];
  const float* W2   = (const float*)d_in[6];
  const float* a2s  = (const float*)d_in[7];
  const float* a2d  = (const float*)d_in[8];
  const float* R2   = (const float*)d_in[9];
  const float* Wd1  = (const float*)d_in[10];
  const float* ad1s = (const float*)d_in[11];
  const float* ad1d = (const float*)d_in[12];
  const float* Rd1  = (const float*)d_in[13];
  const float* Wd2  = (const float*)d_in[14];
  const float* ad2s = (const float*)d_in[15];
  const float* ad2d = (const float*)d_in[16];
  const float* Rd2  = (const float*)d_in[17];
  const float* fc1w = (const float*)d_in[18];
  const float* fc1b = (const float*)d_in[19];
  const float* fc2w = (const float*)d_in[20];
  const float* fc2b = (const float*)d_in[21];
  const float* convw= (const float*)d_in[22];
  const float* convb= (const float*)d_in[23];
  const int*   ei   = (const int*)d_in[24];
  const int* src = ei;
  const int* dst = ei + NEDGES;

  char* p = (char*)d_ws;
  LayerWS w;
  w.xb   = (__bf16*)bump(p, (size_t)NNODES * 256 * sizeof(__bf16));       // max Fin=256
  w.wb   = (__bf16*)bump(p, (size_t)NHEAD * 256 * FHID * sizeof(__bf16));
  w.rb   = (__bf16*)bump(p, (size_t)256 * FHID * sizeof(__bf16));
  w.h    = (float*)bump(p, (size_t)NHEAD * NNODES * FHID * sizeof(float));
  w.res  = (float*)bump(p, (size_t)NNODES * FHID * sizeof(float));
  w.ssrc = (float*)bump(p, (size_t)NNODES * NHEAD * sizeof(float));
  w.sdst = (float*)bump(p, (size_t)NNODES * NHEAD * sizeof(float));
  w.m    = (float*)bump(p, (size_t)NNODES * NHEAD * sizeof(float));
  w.den  = (float*)bump(p, (size_t)NNODES * NHEAD * sizeof(float));
  w.ebuf = (float*)bump(p, (size_t)NEDGES * NHEAD * sizeof(float));
  w.agg  = (float*)bump(p, (size_t)NNODES * FHID * sizeof(float));
  float* x1 = (float*)bump(p, (size_t)NNODES * FHID * sizeof(float));
  float* x2 = (float*)bump(p, (size_t)NNODES * FHID * sizeof(float));
  float* d1 = (float*)bump(p, (size_t)NNODES * FHID * sizeof(float));
  float* d2 = (float*)bump(p, (size_t)NNODES * FHID * sizeof(float));
  float* sums = (float*)bump(p, 4 * sizeof(float));
  float* att  = (float*)bump(p, 4 * sizeof(float));

  // 4 GAT layers
  gat_layer(x,   256, W1,  a1s,  a1d,  R1,  src, dst, x1, w, stream);
  gat_layer(x1,  128, W2,  a2s,  a2d,  R2,  src, dst, x2, w, stream);
  gat_layer(dat, 128, Wd1, ad1s, ad1d, Rd1, src, dst, d1, w, stream);
  gat_layer(d1,  128, Wd2, ad2s, ad2d, Rd2, src, dst, d2, w, stream);

  // SE block + 1x1 conv combine
  size_t total = (size_t)NNODES * FHID;
  fill_f32<<<1, 32, 0, stream>>>(sums, 0.f, 4);
  dim3 gred(240, 4);
  chan_reduce<<<gred, 256, 0, stream>>>(x1, x2, d1, d2, sums, total);
  se_fc<<<1, 32, 0, stream>>>(sums, 1.0f / (float)total, fc1w, fc1b, fc2w, fc2b, att);
  se_combine<<<(unsigned)((total + 255) / 256), 256, 0, stream>>>(
      x1, x2, d1, d2, att, convw, convb, (float*)d_out, total);
}
